// EyeExpertM_54795192762722
// MI455X (gfx1250) — matrix operand order repaired and checked by reading, not compiled, not run on hardware
//
#include <hip/hip_runtime.h>
#include <hip/hip_bf16.h>

typedef __attribute__((ext_vector_type(16))) _Float16 v16h;
typedef __attribute__((ext_vector_type(8)))  float    v8f;

#define B_  128
#define T_  256
#define W_  200
#define H_  256
#define E_  768
#define EMB_ 32
#define IN_ 800        // E + EMB
#define G3_ 768        // 3*H
#define WPAD_ 256      // padded word dim for logits GEMM
#define NEG_ (-1e9f)

// ---------------------------------------------------------------------------
// CDNA5 async global->LDS copy (ASYNCcnt-tracked, no VGPR round trip).
// VDST = per-lane LDS byte address (low 32 bits of generic pointer = wave-
// relative LDS offset per ISA aperture rules), VADDR = 64-bit global address.
// ---------------------------------------------------------------------------
__device__ inline void async_cp16(unsigned lds_addr, const void* gaddr) {
    asm volatile("global_load_async_to_lds_b128 %0, %1, off"
                 :: "v"(lds_addr), "v"(gaddr) : "memory");
}
__device__ inline void wait_async0() {
    asm volatile("s_wait_asynccnt 0x0" ::: "memory");
}

// ---------------------------------------------------------------------------
// WMMA fragment helpers (CDNA5 ISA 7.12.2 layouts, wave32)
// ---------------------------------------------------------------------------
union Frag16 { v16h v; unsigned int u[8]; };

// 16x32 f16 A-matrix fragment from an LDS tile.
__device__ inline v16h afrag_lds(const _Float16* base, int stride, int k0, int lane) {
    int m  = lane & 15;
    int kb = (lane < 16) ? 0 : 16;
    const _Float16* p = base + m * stride + k0 + kb;
    Frag16 f;
#pragma unroll
    for (int i = 0; i < 4; ++i) {
        f.u[i]     = *(const unsigned int*)(p + 2 * i);
        f.u[i + 4] = *(const unsigned int*)(p + 8 + 2 * i);
    }
    return f.v;
}

// 32x16 f16 B-matrix fragment from an LDS tile stored as Bs[n][k] (n-major).
__device__ inline v16h bfrag_lds(const _Float16* base, int stride, int lane) {
    int n  = lane & 15;
    int kb = (lane < 16) ? 0 : 16;
    const _Float16* p = base + n * stride + kb;
    Frag16 f;
#pragma unroll
    for (int i = 0; i < 8; ++i)
        f.u[i] = *(const unsigned int*)(p + 2 * i);
    return f.v;
}

// Same B fragment straight from a row-major weight matrix W[N][K] in global.
__device__ inline v16h bfrag_g(const _Float16* Wm, int ldk, int n0, int k0, int lane) {
    int n  = n0 + (lane & 15);
    int kb = (lane < 16) ? 0 : 16;
    const _Float16* p = Wm + (size_t)n * ldk + k0 + kb;
    Frag16 f;
    uint4 lo = *(const uint4*)(p);
    uint4 hi = *(const uint4*)(p + 8);
    f.u[0] = lo.x; f.u[1] = lo.y; f.u[2] = lo.z; f.u[3] = lo.w;
    f.u[4] = hi.x; f.u[5] = hi.y; f.u[6] = hi.z; f.u[7] = hi.w;
    return f.v;
}

__device__ inline v8f wmma_f16(v16h a, v16h b, v8f c) {
    return __builtin_amdgcn_wmma_f32_16x16x32_f16(false, a, false, b, (short)0, c,
                                                  false, false);
}

__device__ inline float sigmoidf_(float x) { return 1.f / (1.f + __expf(-x)); }

// ---------------------------------------------------------------------------
// Pack kernels
// ---------------------------------------------------------------------------
__global__ void pack_rnn(const float* __restrict__ inputs,
                         const int* __restrict__ fix_seq,
                         const float* __restrict__ scan_emb,
                         _Float16* __restrict__ rnn) {
    int m  = blockIdx.x;                 // 0 .. B*T-1
    int fs = fix_seq[m];
    for (int k = threadIdx.x; k < IN_; k += 256) {
        float v = (k < E_) ? inputs[(size_t)m * E_ + k]
                           : scan_emb[(size_t)fs * EMB_ + (k - E_)];
        rnn[(size_t)m * IN_ + k] = (_Float16)v;
    }
}

__global__ void pack_weights(const float* __restrict__ Wih, const float* __restrict__ Whh,
                             const float* __restrict__ outW,
                             const float* __restrict__ bih, const float* __restrict__ bhh,
                             const int* __restrict__ expert,
                             _Float16* __restrict__ WihH, _Float16* __restrict__ WhhH,
                             _Float16* __restrict__ outWH,
                             float* __restrict__ bihW, float* __restrict__ bhhW) {
    int e = expert[0];
    size_t n1 = (size_t)G3_ * IN_;
    size_t n2 = (size_t)G3_ * H_;
    size_t stride = (size_t)gridDim.x * 256;
    size_t i0 = (size_t)blockIdx.x * 256 + threadIdx.x;
    for (size_t k = i0; k < n1; k += stride) WihH[k] = (_Float16)Wih[(size_t)e * n1 + k];
    for (size_t k = i0; k < n2; k += stride) WhhH[k] = (_Float16)Whh[(size_t)e * n2 + k];
    for (size_t k = i0; k < n2; k += stride) outWH[k] = (_Float16)outW[k];
    for (size_t k = i0; k < G3_; k += stride) {
        bihW[k] = bih[(size_t)e * G3_ + k];
        bhhW[k] = bhh[(size_t)e * G3_ + k];
    }
}

__global__ void pack_fwe(const float* __restrict__ fwe,
                         _Float16* __restrict__ fweP,
                         int* __restrict__ wmask) {
    int gw   = blockIdx.x * 8 + (threadIdx.x >> 5);   // 0 .. B*256-1
    int lane = threadIdx.x & 31;
    int b = gw >> 8, w = gw & 255;
    const float* src = (w < W_) ? fwe + ((size_t)b * W_ + w) * E_ : nullptr;
    _Float16* dst = fweP + ((size_t)b * WPAD_ + w) * E_;
    float s = 0.f;
    for (int e = lane; e < E_; e += 32) {
        float v = src ? src[e] : 0.f;
        s += fabsf(v);
        dst[e] = (_Float16)v;
    }
#pragma unroll
    for (int o = 16; o; o >>= 1) s += __shfl_xor(s, o, 32);
    if (lane == 0) wmask[gw] = (s != 0.f) ? 1 : 0;
}

// ---------------------------------------------------------------------------
// Generic WMMA GEMM:  C[M,N] = A[M,K] @ W[N,K]^T + bias
// Block: 256 threads (8 waves), tile 128(M) x 64(N); waves 4x2, 2x2 WMMA tiles.
// Double-buffered GLOBAL_LOAD_ASYNC_TO_LDS staging.
// ---------------------------------------------------------------------------
#define LSTR 40   // padded LDS stride in halfs (80B: bank-conflict free, 16B aligned)

__device__ inline void stage_async(const _Float16* A, const _Float16* Wm,
                                   _Float16* As, _Float16* Bs,
                                   int m0, int n0, int k0, int K, int tid) {
    int r = tid >> 1, c = (tid & 1) * 16;
    unsigned la = (unsigned)(size_t)(As + r * LSTR + c);
    const _Float16* ga = A + (size_t)(m0 + r) * K + k0 + c;
    async_cp16(la, ga);
    async_cp16(la + 16u, ga + 8);
    if (tid < 128) {
        unsigned lb = (unsigned)(size_t)(Bs + r * LSTR + c);
        const _Float16* gb = Wm + (size_t)(n0 + r) * K + k0 + c;
        async_cp16(lb, gb);
        async_cp16(lb + 16u, gb + 8);
    }
}

__global__ __launch_bounds__(256)
void gemm_f16(const _Float16* __restrict__ A, const _Float16* __restrict__ Wm,
              const float* __restrict__ bias,
              float* __restrict__ Cf, _Float16* __restrict__ Ch,
              int M, int N, int K) {
    __shared__ _Float16 As[2][128 * LSTR];
    __shared__ _Float16 Bs[2][64 * LSTR];
    int tid = threadIdx.x, lane = tid & 31, wid = tid >> 5;
    int wm = wid & 3, wn = wid >> 2;
    int m0 = blockIdx.x * 128, n0 = blockIdx.y * 64;
    v8f acc[2][2] = {};
    const int NK = K / 32;

    stage_async(A, Wm, As[0], Bs[0], m0, n0, 0, K, tid);
    for (int i = 0; i < NK; ++i) {
        wait_async0();          // my async writes for buf i&1 have landed
        __syncthreads();        // everyone's have
        if (i + 1 < NK)
            stage_async(A, Wm, As[(i + 1) & 1], Bs[(i + 1) & 1],
                        m0, n0, (i + 1) * 32, K, tid);
        const _Float16* Ab = As[i & 1];
        const _Float16* Bb = Bs[i & 1];
        v16h a0 = afrag_lds(Ab + (wm * 32) * LSTR,      LSTR, 0, lane);
        v16h a1 = afrag_lds(Ab + (wm * 32 + 16) * LSTR, LSTR, 0, lane);
        v16h b0 = bfrag_lds(Bb + (wn * 32) * LSTR,      LSTR, lane);
        v16h b1 = bfrag_lds(Bb + (wn * 32 + 16) * LSTR, LSTR, lane);
        acc[0][0] = wmma_f16(a0, b0, acc[0][0]);
        acc[0][1] = wmma_f16(a0, b1, acc[0][1]);
        acc[1][0] = wmma_f16(a1, b0, acc[1][0]);
        acc[1][1] = wmma_f16(a1, b1, acc[1][1]);
        __syncthreads();        // all waves done reading buf i&1
    }

    int ml = (lane < 16) ? 0 : 8, nl = lane & 15;
#pragma unroll
    for (int mi = 0; mi < 2; ++mi)
#pragma unroll
        for (int ni = 0; ni < 2; ++ni) {
            int col = n0 + wn * 32 + ni * 16 + nl;
            float bv = bias ? bias[col] : 0.f;
#pragma unroll
            for (int v = 0; v < 8; ++v) {
                int row = m0 + wm * 32 + mi * 16 + ml + v;
                float x = acc[mi][ni][v] + bv;
                if (Cf) Cf[(size_t)row * N + col] = x;
                if (Ch) Ch[(size_t)row * N + col] = (_Float16)x;
            }
        }
}

// ---------------------------------------------------------------------------
// GRU recurrence. One block = 16 batch rows for all 256 time steps.
// Whh held ENTIRELY in VGPRs (48 B-fragments/wave = 384 VGPRs; gfx1250 wave32
// has a 1024-VGPR budget via VGPR-MSB). h kept in LDS (f32 state + f16 A-tile).
// Per step: 8 K-slices x 6 N-tiles/wave of v_wmma_f32_16x16x32_f16.
// ---------------------------------------------------------------------------
#define HAS 264   // hA stride (halfs):  528B rows -> 16B aligned, conflict-free
#define HGS 776   // hg stride (halfs)

__global__ __launch_bounds__(256, 1)
void gru_kernel(const float* __restrict__ xg, const _Float16* __restrict__ Whh,
                const float* __restrict__ bhh, const int* __restrict__ lengths,
                float* __restrict__ hOutF, _Float16* __restrict__ hOutH) {
    __shared__ _Float16 hA[16 * HAS];
    __shared__ _Float16 hg[16 * HGS];
    __shared__ float    hF[16 * H_];
    __shared__ int      llen[16];

    int tid = threadIdx.x, lane = tid & 31, wid = tid >> 5;
    int b0 = blockIdx.x * 16;

    for (int i = tid; i < 16 * HAS; i += 256) hA[i] = (_Float16)0.f;
    for (int i = tid; i < 16 * H_;  i += 256) hF[i] = 0.f;
    if (tid < 16) llen[tid] = lengths[b0 + tid];

    // preload all Whh fragments for this wave's 6 N-tiles x 8 K-slices
    v16h wf[8][6];
#pragma unroll
    for (int ks = 0; ks < 8; ++ks)
#pragma unroll
        for (int j = 0; j < 6; ++j)
            wf[ks][j] = bfrag_g(Whh, H_, (wid * 6 + j) * 16, ks * 32, lane);

    __syncthreads();

    int ml = (lane < 16) ? 0 : 8, nl = lane & 15;

    for (int t = 0; t < T_; ++t) {
        v8f acc[6] = {};
#pragma unroll
        for (int ks = 0; ks < 8; ++ks) {            // K = 256 in slices of 32
            v16h a = afrag_lds(hA, HAS, ks * 32, lane);
#pragma unroll
            for (int j = 0; j < 6; ++j)
                acc[j] = wmma_f16(a, wf[ks][j], acc[j]);
        }
        // hg = acc + bhh  (f16 in LDS)
#pragma unroll
        for (int j = 0; j < 6; ++j) {
            int n = (wid * 6 + j) * 16 + nl;
            float bv = bhh[n];
#pragma unroll
            for (int v = 0; v < 8; ++v)
                hg[(ml + v) * HGS + n] = (_Float16)(acc[j][v] + bv);
        }
        __syncthreads();

        // gate math: thread j handles hidden index j for all 16 rows
        int j = tid;
        for (int r = 0; r < 16; ++r) {
            const float* xp = xg + ((size_t)(b0 + r) * T_ + t) * G3_;
            float hr = (float)hg[r * HGS + j];
            float hz = (float)hg[r * HGS + j + H_];
            float hn = (float)hg[r * HGS + j + 2 * H_];
            float rg = sigmoidf_(xp[j] + hr);
            float zg = sigmoidf_(xp[j + H_] + hz);
            float ng = tanhf(xp[j + 2 * H_] + rg * hn);
            float hp = hF[r * H_ + j];
            float hv = (1.f - zg) * ng + zg * hp;
            hF[r * H_ + j] = hv;
            hA[r * HAS + j] = (_Float16)hv;
            float ov = (t < llen[r]) ? hv : 0.f;
            size_t o = ((size_t)(b0 + r) * T_ + t) * H_ + j;
            hOutF[o] = ov;
            hOutH[o] = (_Float16)ov;
        }
        __syncthreads();
    }
}

// ---------------------------------------------------------------------------
// Per-batch logits GEMM: logits[b] = proj[b] @ fwe[b]^T with mask epilogue.
// Grid: (T/128, WPAD/64, B). Same async double-buffered tiling as gemm_f16.
// ---------------------------------------------------------------------------
__global__ __launch_bounds__(256)
void logits_kernel(const _Float16* __restrict__ proj, const _Float16* __restrict__ fweP,
                   const int* __restrict__ fix_seq, const int* __restrict__ wmask,
                   float* __restrict__ logits) {
    __shared__ _Float16 As[2][128 * LSTR];
    __shared__ _Float16 Bs[2][64 * LSTR];
    int tid = threadIdx.x, lane = tid & 31, wid = tid >> 5;
    int wm = wid & 3, wn = wid >> 2;
    int bb = blockIdx.z;
    int m0 = blockIdx.x * 128, n0 = blockIdx.y * 64;
    const _Float16* A  = proj + (size_t)bb * T_ * E_;
    const _Float16* Wm = fweP + (size_t)bb * WPAD_ * E_;
    v8f acc[2][2] = {};
    const int NK = E_ / 32;

    stage_async(A, Wm, As[0], Bs[0], m0, n0, 0, E_, tid);
    for (int i = 0; i < NK; ++i) {
        wait_async0();
        __syncthreads();
        if (i + 1 < NK)
            stage_async(A, Wm, As[(i + 1) & 1], Bs[(i + 1) & 1],
                        m0, n0, (i + 1) * 32, E_, tid);
        const _Float16* Ab = As[i & 1];
        const _Float16* Bb = Bs[i & 1];
        v16h a0 = afrag_lds(Ab + (wm * 32) * LSTR,      LSTR, 0, lane);
        v16h a1 = afrag_lds(Ab + (wm * 32 + 16) * LSTR, LSTR, 0, lane);
        v16h b0 = bfrag_lds(Bb + (wn * 32) * LSTR,      LSTR, lane);
        v16h b1 = bfrag_lds(Bb + (wn * 32 + 16) * LSTR, LSTR, lane);
        acc[0][0] = wmma_f16(a0, b0, acc[0][0]);
        acc[0][1] = wmma_f16(a0, b1, acc[0][1]);
        acc[1][0] = wmma_f16(a1, b0, acc[1][0]);
        acc[1][1] = wmma_f16(a1, b1, acc[1][1]);
        __syncthreads();
    }

    int ml = (lane < 16) ? 0 : 8, nl = lane & 15;
#pragma unroll
    for (int mi = 0; mi < 2; ++mi)
#pragma unroll
        for (int ni = 0; ni < 2; ++ni) {
            int w = n0 + wn * 32 + ni * 16 + nl;
            if (w >= W_) continue;
            int wm_ok = wmask[bb * WPAD_ + w];
#pragma unroll
            for (int v = 0; v < 8; ++v) {
                int t = m0 + wm * 32 + mi * 16 + ml + v;
                int fs = fix_seq[bb * T_ + t];
                int d = w - fs; if (d < 0) d = -d;
                float val = (d <= 8 && wm_ok) ? acc[mi][ni][v] : NEG_;
                logits[((size_t)bb * T_ + t) * W_ + w] = val;
            }
        }
}

// ---------------------------------------------------------------------------
// dur_pred[b,t] = dot(outputs[b,t,:], dur_W) + dur_b   (one wave per row)
// ---------------------------------------------------------------------------
__global__ void dur_kernel(const float* __restrict__ outs,
                           const float* __restrict__ durW,
                           const float* __restrict__ durB,
                           float* __restrict__ out) {
    int gw   = blockIdx.x * 8 + (threadIdx.x >> 5);
    int lane = threadIdx.x & 31;
    const float* p = outs + (size_t)gw * H_;
    float s = 0.f;
#pragma unroll
    for (int h = lane; h < H_; h += 32) s += p[h] * durW[h];
#pragma unroll
    for (int o = 16; o; o >>= 1) s += __shfl_xor(s, o, 32);
    if (lane == 0) out[gw] = s + durB[0];
}

// ---------------------------------------------------------------------------
extern "C" void kernel_launch(void* const* d_in, const int* in_sizes, int n_in,
                              void* d_out, int out_size, void* d_ws, size_t ws_size,
                              hipStream_t stream) {
    const float* inputs   = (const float*)d_in[0];
    const int*   fix_seq  = (const int*)d_in[1];
    const float* fwe      = (const float*)d_in[2];
    const int*   lengths  = (const int*)d_in[3];
    const int*   expert   = (const int*)d_in[4];
    const float* scan_emb = (const float*)d_in[5];
    const float* gWih     = (const float*)d_in[6];
    const float* gWhh     = (const float*)d_in[7];
    const float* gbih     = (const float*)d_in[8];
    const float* gbhh     = (const float*)d_in[9];
    const float* outW     = (const float*)d_in[10];
    const float* outB     = (const float*)d_in[11];
    const float* durW     = (const float*)d_in[12];
    const float* durB     = (const float*)d_in[13];
    float* out = (float*)d_out;

    char* base = (char*)d_ws;
    size_t off = 0;
    auto alloc = [&](size_t bytes) -> char* {
        char* p = base + off;
        off = (off + bytes + 255) & ~(size_t)255;
        return p;
    };
    const size_t MT = (size_t)B_ * T_;                       // 32768
    _Float16* rnnH  = (_Float16*)alloc(MT * IN_ * 2);        // 52.4 MB
    _Float16* WihH  = (_Float16*)alloc((size_t)G3_ * IN_ * 2);
    _Float16* WhhH  = (_Float16*)alloc((size_t)G3_ * H_ * 2);
    _Float16* outWH = (_Float16*)alloc((size_t)G3_ * H_ * 2);
    _Float16* fweP  = (_Float16*)alloc((size_t)B_ * WPAD_ * E_ * 2);  // 50.3 MB
    int*      wmask = (int*)alloc((size_t)B_ * WPAD_ * 4);
    float*    bihW  = (float*)alloc((size_t)G3_ * 4);
    float*    bhhW  = (float*)alloc((size_t)G3_ * 4);
    float*    xg    = (float*)alloc(MT * G3_ * 4);           // 100.7 MB
    float*    hOutF = (float*)alloc(MT * H_ * 4);            // 33.6 MB
    _Float16* hOutH = (_Float16*)alloc(MT * H_ * 2);
    _Float16* projH = (_Float16*)alloc(MT * E_ * 2);         // 50.3 MB
    (void)ws_size; (void)in_sizes; (void)n_in; (void)out_size;

    float* logits_out = out;                                 // [B,T,200]
    float* dur_out    = out + MT * W_;                       // [B,T]

    pack_weights<<<512, 256, 0, stream>>>(gWih, gWhh, outW, gbih, gbhh, expert,
                                          WihH, WhhH, outWH, bihW, bhhW);
    pack_rnn<<<(int)MT, 256, 0, stream>>>(inputs, fix_seq, scan_emb, rnnH);
    pack_fwe<<<(B_ * WPAD_) / 8, 256, 0, stream>>>(fwe, fweP, wmask);

    gemm_f16<<<dim3(MT / 128, G3_ / 64), 256, 0, stream>>>(
        rnnH, WihH, bihW, xg, nullptr, (int)MT, G3_, IN_);

    gru_kernel<<<B_ / 16, 256, 0, stream>>>(xg, WhhH, bhhW, lengths, hOutF, hOutH);

    gemm_f16<<<dim3(MT / 128, G3_ / 64), 256, 0, stream>>>(
        hOutH, outWH, outB, nullptr, projH, (int)MT, G3_, H_);

    logits_kernel<<<dim3(T_ / 128, WPAD_ / 64, B_), 256, 0, stream>>>(
        projH, fweP, fix_seq, wmask, logits_out);

    dur_kernel<<<(int)(MT / 8), 256, 0, stream>>>(hOutF, durW, durB, dur_out);
}
